// TERMGraphGVPEncoder_79817672229198
// MI455X (gfx1250) — matrix operand order repaired and checked by reading, not compiled, not run on hardware
//
#include <hip/hip_runtime.h>

// ---------------------------------------------------------------------------
// TERM Graph GVP encoder for gfx1250 (MI455X), bf16 WMMA w/ fp32 accumulate.
//
// Problem dims (fixed by the reference):
#define NV_   64
#define NS_   64
#define HIDN  128
#define B_    2
#define T_    32
#define N_    24
#define K_    24
#define NODES 1536                 // B*T*N
#define EDGES 36864                // NODES*K
#define REP   256                  // merged rep width: 3*NV + NS
#define VLEN  192                  // 3*NV
#define EPSF  1e-8f

// Assumed d_in flattening (compile-only environment; documented assumption):
//  top-level dict insertion order: V, E, E_idx, mask, params
//  params flattened as a JAX pytree (dict keys sorted alphabetically):
//   [4..7]   W_e   (bs, wh, ws, wv)
//   [8..11]  W_out (bs, wh, ws, wv)
//   [12..15] W_v   (bs, wh, ws, wv)
//   [16+24l] edge_layers[l]: dh0(bs,wh,ws,wv) dh1 ev0 ev1 ev2 norm0(b,g) norm1(b,g)
//   [88+24l] node_layers[l]: same sub-order
// ---------------------------------------------------------------------------

typedef __attribute__((ext_vector_type(16))) __bf16 v16bf;
typedef __attribute__((ext_vector_type(8)))  float  v8f;

union BFV {
  v16bf v;
  unsigned short u[16];
  uint4 q[2];
};

__device__ inline unsigned short f2bf(float f) {
  unsigned int b = __float_as_uint(f);
  return (unsigned short)((b + 0x7FFFu + ((b >> 16) & 1u)) >> 16);   // RNE
}

__device__ inline v8f wmma_bf16(v16bf a, v16bf b, v8f c) {
  // D(16x16,f32) = A(16x32,bf16) x B(32x16,bf16) + C
  return __builtin_amdgcn_wmma_f32_16x16x32_bf16(false, a, false, b,
                                                 (short)0, c, false, false);
}

__device__ inline float sigm(float v) { return 1.f / (1.f + expf(-v)); }

// ---------------------------------------------------------------------------
// Fused GVP kernel arguments (shape comes from the template instantiation).
// ---------------------------------------------------------------------------
struct GvpArgs {
  const float* x;                      // amode 0 source [T, 3*vi+si]
  const float* hV;                     // gather source, rows of width REP
  const float* hE;                     // gather source, rows of width REP
  const int*   eidx;                   // [NODES, K]
  const unsigned short *wh, *ws, *wv;  // bf16 weights, fragment-swizzled
  const float* bs;                     // [so] fp32 bias
  float* y;                            // [T, 3*vo+so]
  int ldx, ldy;
  int nl;                              // nonlinear flag
  int amode;                           // 0 plain, 1 node-cat, 2 edge-cat
};

// Fetch one element of the (virtual) A matrix. c>=0: vector part coord c of
// channel `col`; c<0: scalar channel `col`. amode 1/2 fuse the
// cat_term_edge_endpoints gather (own / i=E_idx[..,0] / j=E_idx[..,k] / h_E).
template <int VI>
__device__ inline float fetch_A(const GvpArgs& g, int tkn, int col, int c) {
  if (g.amode == 0) {
    const float* row = g.x + (size_t)tkn * g.ldx;
    return (c >= 0) ? row[col * 3 + c] : row[3 * VI + col];
  }
  int gi  = tkn / K_;                  // graph node linear index
  int cb  = (gi / N_) * N_;            // cell base
  int sel = (col >> 6) + ((g.amode == 2) ? 1 : 0);
  int ch  = col & 63;
  const float* row;
  if (sel == 0)      row = g.hV + (size_t)gi * REP;
  else if (sel == 1) row = g.hV + (size_t)(cb + g.eidx[(size_t)gi * K_]) * REP;
  else if (sel == 2) row = g.hV + (size_t)(cb + g.eidx[tkn]) * REP;
  else               row = g.hE + (size_t)tkn * REP;
  return (c >= 0) ? row[ch * 3 + c] : row[VLEN + ch];
}

// A fragment (16x32 bf16) from a bf16 LDS tile, row stride 32.
// ISA layout: lane<16 row m=lane holds K {0..7}U{16..23}; lane>=16 {8..15}U{24..31}
__device__ inline v16bf frag_a_bf(const unsigned short* s, int rowBase,
                                  int lane) {
  int m  = rowBase + (lane & 15);
  int kb = (lane < 16) ? 0 : 8;
  const unsigned short* r = s + m * 32 + kb;
  BFV t;
  t.q[0] = *reinterpret_cast<const uint4*>(r);        // K kb..kb+7
  t.q[1] = *reinterpret_cast<const uint4*>(r + 16);   // K kb+16..kb+23
  return t.v;
}

// A fragment from an fp32 LDS tile (row stride LD), converted in-register.
template <int LD>
__device__ inline v16bf frag_a_f32(const float* s, int rowBase, int kofs,
                                   int lane) {
  int m  = rowBase + (lane & 15);
  int kb = kofs + ((lane < 16) ? 0 : 8);
  const float* r = s + m * LD + kb;
  float4 x0 = *reinterpret_cast<const float4*>(r);
  float4 x1 = *reinterpret_cast<const float4*>(r + 4);
  float4 x2 = *reinterpret_cast<const float4*>(r + 16);
  float4 x3 = *reinterpret_cast<const float4*>(r + 20);
  BFV t;
  t.u[0]  = f2bf(x0.x); t.u[1]  = f2bf(x0.y);
  t.u[2]  = f2bf(x0.z); t.u[3]  = f2bf(x0.w);
  t.u[4]  = f2bf(x1.x); t.u[5]  = f2bf(x1.y);
  t.u[6]  = f2bf(x1.z); t.u[7]  = f2bf(x1.w);
  t.u[8]  = f2bf(x2.x); t.u[9]  = f2bf(x2.y);
  t.u[10] = f2bf(x2.z); t.u[11] = f2bf(x2.w);
  t.u[12] = f2bf(x3.x); t.u[13] = f2bf(x3.y);
  t.u[14] = f2bf(x3.z); t.u[15] = f2bf(x3.w);
  return t.v;
}

// B fragment from fragment-swizzled weights: tile (ktile, ct) of an
// [Kdim, 16*NCT] matrix stored as [(ktile*NCT+ct)][lane][e] contiguous.
template <int NCT>
__device__ inline v16bf frag_b_swz(const unsigned short* w, int ktile, int ct,
                                   int lane) {
  const unsigned short* r =
      w + (((size_t)(ktile * NCT + ct) * 32) + lane) * 16;
  BFV t;
  t.q[0] = *reinterpret_cast<const uint4*>(r);
  t.q[1] = *reinterpret_cast<const uint4*>(r + 8);
  return t.v;
}

// Store 16x16 f32 accumulator tile into LDS (C/D layout: VGPR r -> m=r(+8)).
template <int LD>
__device__ inline void acc_to_lds(float* s, int rowBase, int colBase, int lane,
                                  v8f acc) {
  int n  = colBase + (lane & 15);
  int mo = rowBase + ((lane < 16) ? 0 : 8);
#pragma unroll
  for (int r = 0; r < 8; ++r) s[(mo + r) * LD + n] = acc[r];
}

// ---------------------------------------------------------------------------
// Fused GVP: one workgroup = 16 tokens; vh/vn/vu live in LDS.
//   GEMM1: vh[48,H]  = Avec[48,VI] x wh[VI,H]       (rows = (token,coord))
//   vn[16,H] = ||vh|| over coords
//   GEMM2: out[16,SO]= [S | vn][16,SI+H] x ws       (+bias, sigmoid if nl)
//   GEMM3: vu[48,VO] = vh x wv                      (vector gate if nl)
// ---------------------------------------------------------------------------
template <int VI, int SI, int VO, int SO>
__global__ void __launch_bounds__(256) gvp_kernel(GvpArgs g) {
  constexpr int H    = (VI > VO) ? VI : VO;
  constexpr int NCT1 = H / 16,  NT1 = 3 * NCT1, L1 = (NT1 + 7) / 8;
  constexpr int NCT2 = SO / 16, L2 = (NCT2 + 7) / 8;
  constexpr int NCT3 = VO / 16, NT3 = 3 * NCT3, L3 = (NT3 + 7) / 8;
  constexpr int K2   = SI + H;

  extern __shared__ char smemraw[];
  float* vh = reinterpret_cast<float*>(smemraw);          // 48*H f32
  float* vn = vh + 48 * H;                                // 16*H f32
  float* vu = vn + 16 * H;                                // 48*VO f32
  unsigned short* As =
      reinterpret_cast<unsigned short*>(vu + 48 * VO);    // 48*32 bf16

  const int tid  = threadIdx.x;
  const int wave = (tid >> 5) & 7;
  const int lane = tid & 31;
  const int tok0 = blockIdx.x * 16;

  const v8f zero = {0.f, 0.f, 0.f, 0.f, 0.f, 0.f, 0.f, 0.f};

  // ---- GEMM1: vh = Avec x wh ---------------------------------------------
  v8f acc1[L1];
#pragma unroll
  for (int i = 0; i < L1; ++i) acc1[i] = zero;

  for (int kt = 0; kt < VI; kt += 32) {
    for (int i = tid; i < 48 * 32; i += 256) {
      int row = i >> 5, kk = i & 31;
      As[i] = f2bf(fetch_A<VI>(g, tok0 + row / 3, kt + kk, row % 3));
    }
    __syncthreads();
#pragma unroll
    for (int li = 0; li < L1; ++li) {
      int tile = wave + li * 8;
      if (tile < NT1) {
        int rt = tile % 3, ct = tile / 3;
        v16bf a = frag_a_bf(As, rt * 16, lane);
        v16bf b = frag_b_swz<NCT1>(g.wh, kt >> 5, ct, lane);
        acc1[li] = wmma_bf16(a, b, acc1[li]);
      }
    }
    __syncthreads();
  }
#pragma unroll
  for (int li = 0; li < L1; ++li) {
    int tile = wave + li * 8;
    if (tile < NT1) {
      int rt = tile % 3, ct = tile / 3;
      acc_to_lds<H>(vh, rt * 16, ct * 16, lane, acc1[li]);
    }
  }
  __syncthreads();

  // ---- vn = ||vh|| over coords -------------------------------------------
  for (int i = tid; i < 16 * H; i += 256) {
    int t = i / H, hh = i - t * H;
    float a0 = vh[(t * 3 + 0) * H + hh];
    float a1 = vh[(t * 3 + 1) * H + hh];
    float a2 = vh[(t * 3 + 2) * H + hh];
    vn[i] = sqrtf(fmaxf(a0 * a0 + a1 * a1 + a2 * a2, EPSF));
  }
  __syncthreads();

  // ---- GEMM2: scalar head -------------------------------------------------
  v8f acc2[L2];
#pragma unroll
  for (int i = 0; i < L2; ++i) acc2[i] = zero;

  for (int kt = 0; kt < K2; kt += 32) {
    for (int i = tid; i < 16 * 32; i += 256) {
      int t = i >> 5, kk = kt + (i & 31);
      float val = (kk < SI) ? fetch_A<VI>(g, tok0 + t, kk, -1)
                            : vn[t * H + (kk - SI)];
      As[i] = f2bf(val);
    }
    __syncthreads();
#pragma unroll
    for (int li = 0; li < L2; ++li) {
      int tile = wave + li * 8;
      if (tile < NCT2) {
        v16bf a = frag_a_bf(As, 0, lane);
        v16bf b = frag_b_swz<NCT2>(g.ws, kt >> 5, tile, lane);
        acc2[li] = wmma_bf16(a, b, acc2[li]);
      }
    }
    __syncthreads();
  }
#pragma unroll
  for (int li = 0; li < L2; ++li) {
    int tile = wave + li * 8;
    if (tile < NCT2) {
      int n  = tile * 16 + (lane & 15);
      int mo = (lane < 16) ? 0 : 8;
      float bias = g.bs[n];
      v8f a = acc2[li];
#pragma unroll
      for (int r = 0; r < 8; ++r) {
        float v = a[r] + bias;
        if (g.nl) v = sigm(v);
        g.y[(size_t)(tok0 + mo + r) * g.ldy + 3 * VO + n] = v;
      }
    }
  }

  // ---- GEMM3: vector head -------------------------------------------------
  v8f acc3[L3];
#pragma unroll
  for (int i = 0; i < L3; ++i) acc3[i] = zero;

  for (int kt = 0; kt < H; kt += 32) {
#pragma unroll
    for (int li = 0; li < L3; ++li) {
      int tile = wave + li * 8;
      if (tile < NT3) {
        int rt = tile % 3, ct = tile / 3;
        v16bf a = frag_a_f32<H>(vh, rt * 16, kt, lane);
        v16bf b = frag_b_swz<NCT3>(g.wv, kt >> 5, ct, lane);
        acc3[li] = wmma_bf16(a, b, acc3[li]);
      }
    }
  }
#pragma unroll
  for (int li = 0; li < L3; ++li) {
    int tile = wave + li * 8;
    if (tile < NT3) {
      int rt = tile % 3, ct = tile / 3;
      acc_to_lds<VO>(vu, rt * 16, ct * 16, lane, acc3[li]);
    }
  }
  __syncthreads();

  for (int i = tid; i < 16 * VO; i += 256) {
    int t = i / VO, o = i - t * VO;
    float u0 = vu[(t * 3 + 0) * VO + o];
    float u1 = vu[(t * 3 + 1) * VO + o];
    float u2 = vu[(t * 3 + 2) * VO + o];
    if (g.nl) {
      float nrm = sqrtf(fmaxf(u0 * u0 + u1 * u1 + u2 * u2, EPSF));
      float sg  = sigm(nrm);
      u0 *= sg; u1 *= sg; u2 *= sg;
    }
    float* yr = g.y + (size_t)(tok0 + t) * g.ldy;
    yr[o * 3 + 0] = u0; yr[o * 3 + 1] = u1; yr[o * 3 + 2] = u2;
  }
}

// ---------------------------------------------------------------------------
// Residual + (masked mean) + GVP layernorm, one 128-thread WG per token.
// deltaMode: 0 plain add of m[tok]; 1 mean over K of m[tok*K+k]*att;
//            2 m[tok]*att (edge).  outMask: 0 none, 1 node mask, 2 edge att.
// ---------------------------------------------------------------------------
__device__ inline float blksum128(float v, float* red, int tid) {
  __syncthreads();
  red[tid] = v;
  __syncthreads();
  for (int s = 64; s >= 1; s >>= 1) {
    if (tid < s) red[tid] += red[tid + s];
    __syncthreads();
  }
  return red[0];
}

__global__ void __launch_bounds__(128) ln_kernel(
    float* x, const float* m, const float* maskV, const int* eidx,
    const float* gg, const float* bb, int deltaMode, int outMask) {
  __shared__ float xs[REP];
  __shared__ float red[128];
  const int tok = blockIdx.x;
  const int tid = threadIdx.x;

  for (int d = tid; d < REP; d += 128) {
    float delta;
    if (deltaMode == 0) {
      delta = m[(size_t)tok * REP + d];
    } else if (deltaMode == 2) {
      int gi = tok / K_;
      int cb = (gi / N_) * N_;
      float att = maskV[gi] * maskV[cb + eidx[tok]];
      delta = m[(size_t)tok * REP + d] * att;
    } else {  // node: masked mean over K neighbors
      int cb = (tok / N_) * N_;
      float mv = maskV[tok];
      float acc = 0.f;
      for (int k = 0; k < K_; ++k) {
        float att = mv * maskV[cb + eidx[(size_t)tok * K_ + k]];
        acc += m[((size_t)tok * K_ + k) * REP + d] * att;
      }
      delta = acc / (float)K_;
    }
    xs[d] = x[(size_t)tok * REP + d] + delta;
  }
  __syncthreads();

  float c0 = 0.f, c1 = 0.f, c2 = 0.f;
  if (tid < NV_) {
    float a = xs[tid * 3], b = xs[tid * 3 + 1], c = xs[tid * 3 + 2];
    c0 = fmaxf(a * a + b * b + c * c, EPSF);
  }
  if (tid < NS_) {
    float sv = xs[VLEN + tid];
    c1 = sv; c2 = sv * sv;
  }
  float vnorm = sqrtf(blksum128(c0, red, tid) / (float)NV_);
  float mu    = blksum128(c1, red, tid) / (float)NS_;
  float var   = fmaxf(blksum128(c2, red, tid) / (float)NS_ - mu * mu, 0.f);
  float rstd  = rsqrtf(var + 1e-5f);

  float om = 1.f;
  if (outMask == 1) {
    om = maskV[tok];
  } else if (outMask == 2) {
    int gi = tok / K_;
    int cb = (gi / N_) * N_;
    om = maskV[gi] * maskV[cb + eidx[tok]];
  }
  for (int d = tid; d < REP; d += 128) {
    float v;
    if (d < VLEN) v = xs[d] / vnorm;
    else {
      int s = d - VLEN;
      v = (xs[d] - mu) * rstd * gg[s] + bb[s];
    }
    x[(size_t)tok * REP + d] = v * om;
  }
}

// Convert fp32 weight [Kdim, Ndim] to bf16, swizzled fragment-major so a
// B fragment is two contiguous 16B loads per lane.
__global__ void __launch_bounds__(256) conv_swz(const float* src,
                                                unsigned short* dst,
                                                int Kdim, int Ndim) {
  int i = blockIdx.x * 256 + threadIdx.x;
  if (i >= Kdim * Ndim) return;
  int e    = i & 15;
  int lane = (i >> 4) & 31;
  int tile = i >> 9;
  int nct  = Ndim >> 4;
  int ct = tile % nct, ktile = tile / nct;
  int n = ct * 16 + (lane & 15);
  int k = ktile * 32 + ((lane < 16) ? 0 : 8) + ((e < 8) ? e : e + 8);
  dst[i] = f2bf(src[(size_t)k * Ndim + n]);
}

// ---------------------------------------------------------------------------
// Host orchestration
// ---------------------------------------------------------------------------
struct GvpH {
  const float *bs, *wh, *ws, *wv;
  unsigned short *whb, *wsb, *wvb;
  int vi, si, h, vo, so;
};

static void gvp_convert(const GvpH& g, hipStream_t s) {
  int n1 = g.vi * g.h, n2 = (g.si + g.h) * g.so, n3 = g.h * g.vo;
  conv_swz<<<(n1 + 255) / 256, 256, 0, s>>>(g.wh, g.whb, g.vi, g.h);
  conv_swz<<<(n2 + 255) / 256, 256, 0, s>>>(g.ws, g.wsb, g.si + g.h, g.so);
  conv_swz<<<(n3 + 255) / 256, 256, 0, s>>>(g.wv, g.wvb, g.h, g.vo);
}

static void gvp_run(const GvpH& g, int T, int amode, const float* x, int ldx,
                    const float* hV, const float* hE, const int* eidx,
                    float* y, int ldy, int nl, hipStream_t s) {
  GvpArgs a;
  a.x = x; a.hV = hV; a.hE = hE; a.eidx = eidx;
  a.wh = g.whb; a.ws = g.wsb; a.wv = g.wvb; a.bs = g.bs;
  a.y = y; a.ldx = ldx; a.ldy = ldy; a.nl = nl; a.amode = amode;

#define GVP_LAUNCH(VI_, SI_, VO_, SO_)                                        \
  do {                                                                        \
    constexpr int H_ = (VI_ > VO_) ? VI_ : VO_;                               \
    size_t sh = sizeof(float) * (48 * H_ + 16 * H_ + 48 * VO_) +              \
                sizeof(unsigned short) * 48 * 32;                             \
    gvp_kernel<VI_, SI_, VO_, SO_><<<T / 16, 256, sh, s>>>(a);                \
  } while (0)

  if      (g.vi == 64  && g.si == 192)               GVP_LAUNCH(64, 192, 64, 64);
  else if (g.vi == 64  && g.si == 64 && g.vo == 64)  GVP_LAUNCH(64, 64, 64, 64);
  else if (g.vi == 256)                              GVP_LAUNCH(256, 256, 64, 64);
  else if (g.vi == 192)                              GVP_LAUNCH(192, 192, 64, 64);
  else if (g.vi == 64  && g.vo == 128)               GVP_LAUNCH(64, 64, 128, 256);
  else                                               GVP_LAUNCH(128, 256, 64, 64);
#undef GVP_LAUNCH
}

extern "C" void kernel_launch(void* const* d_in, const int* in_sizes, int n_in,
                              void* d_out, int out_size, void* d_ws,
                              size_t ws_size, hipStream_t stream) {
  (void)in_sizes; (void)n_in; (void)out_size; (void)ws_size;

  const float* V   = (const float*)d_in[0];
  const float* E   = (const float*)d_in[1];
  const int*   EIX = (const int*)d_in[2];
  const float* MSK = (const float*)d_in[3];

  // ---- bump allocator over d_ws ----
  char* wp = (char*)d_ws;
  auto alloc = [&](size_t bytes) -> void* {
    void* r = (void*)wp;
    wp += (bytes + 255) & ~(size_t)255;
    return r;
  };

  float* hV = (float*)alloc((size_t)NODES * REP * sizeof(float));
  float* hE = (float*)alloc((size_t)EDGES * REP * sizeof(float));
  // message/dh pool: m-chain (2x [EDGES,256]) and dh-chain ([*,640]+[*,256])
  // are temporally disjoint and share this region.
  float* mPool = (float*)alloc((size_t)EDGES * (640 + REP) * sizeof(float));
  float* mA  = mPool;
  float* mB  = mPool + (size_t)EDGES * REP;
  float* dhT = mPool;                            // [T, 640]
  float* dhO = mPool + (size_t)EDGES * 640;      // [T, 256]

  auto mkgvp = [&](int base, int vi, int si, int vo, int so) -> GvpH {
    GvpH g;
    g.bs = (const float*)d_in[base + 0];
    g.wh = (const float*)d_in[base + 1];
    g.ws = (const float*)d_in[base + 2];
    g.wv = (const float*)d_in[base + 3];
    g.vi = vi; g.si = si; g.vo = vo; g.so = so;
    g.h  = (vi > vo) ? vi : vo;
    g.whb = (unsigned short*)alloc((size_t)vi * g.h * 2);
    g.wsb = (unsigned short*)alloc((size_t)(si + g.h) * so * 2);
    g.wvb = (unsigned short*)alloc((size_t)g.h * vo * 2);
    return g;
  };

  GvpH We   = mkgvp(4,  NV_, NS_,        NV_, NS_);
  GvpH Wout = mkgvp(8,  NV_, NS_,        NV_, NS_);
  GvpH Wv   = mkgvp(12, NV_, NS_ + HIDN, NV_, NS_);

  struct LayerH { GvpH dh0, dh1, ev0, ev1, ev2;
                  const float *n0b, *n0g, *n1b, *n1g; };
  LayerH el[3], nlr[3];
  for (int l = 0; l < 3; ++l) {
    int b = 16 + 24 * l;
    el[l].dh0 = mkgvp(b + 0,  NV_,     NS_,     2 * NV_, 4 * NS_);
    el[l].dh1 = mkgvp(b + 4,  2 * NV_, 4 * NS_, NV_,     NS_);
    el[l].ev0 = mkgvp(b + 8,  3 * NV_, 3 * NS_, NV_,     NS_);
    el[l].ev1 = mkgvp(b + 12, NV_,     NS_,     NV_,     NS_);
    el[l].ev2 = mkgvp(b + 16, NV_,     NS_,     NV_,     NS_);
    el[l].n0b = (const float*)d_in[b + 20];
    el[l].n0g = (const float*)d_in[b + 21];
    el[l].n1b = (const float*)d_in[b + 22];
    el[l].n1g = (const float*)d_in[b + 23];
  }
  for (int l = 0; l < 3; ++l) {
    int b = 88 + 24 * l;
    nlr[l].dh0 = mkgvp(b + 0,  NV_,     NS_,     2 * NV_, 4 * NS_);
    nlr[l].dh1 = mkgvp(b + 4,  2 * NV_, 4 * NS_, NV_,     NS_);
    nlr[l].ev0 = mkgvp(b + 8,  4 * NV_, 4 * NS_, NV_,     NS_);
    nlr[l].ev1 = mkgvp(b + 12, NV_,     NS_,     NV_,     NS_);
    nlr[l].ev2 = mkgvp(b + 16, NV_,     NS_,     NV_,     NS_);
    nlr[l].n0b = (const float*)d_in[b + 20];
    nlr[l].n0g = (const float*)d_in[b + 21];
    nlr[l].n1b = (const float*)d_in[b + 22];
    nlr[l].n1g = (const float*)d_in[b + 23];
  }

  // ---- weight conversion + swizzle (every call; deterministic) ----
  gvp_convert(We, stream); gvp_convert(Wout, stream); gvp_convert(Wv, stream);
  for (int l = 0; l < 3; ++l) {
    gvp_convert(el[l].dh0, stream);  gvp_convert(el[l].dh1, stream);
    gvp_convert(el[l].ev0, stream);  gvp_convert(el[l].ev1, stream);
    gvp_convert(el[l].ev2, stream);
    gvp_convert(nlr[l].dh0, stream); gvp_convert(nlr[l].dh1, stream);
    gvp_convert(nlr[l].ev0, stream); gvp_convert(nlr[l].ev1, stream);
    gvp_convert(nlr[l].ev2, stream);
  }

  // ---- pipeline ----
  gvp_run(Wv, NODES, 0, V, 3 * NV_ + NS_ + HIDN, nullptr, nullptr, nullptr,
          hV, REP, 0, stream);
  gvp_run(We, EDGES, 0, E, REP, nullptr, nullptr, nullptr,
          hE, REP, 0, stream);

  for (int l = 0; l < 3; ++l) {
    // ---- node layer ----
    gvp_run(nlr[l].ev0, EDGES, 1, nullptr, 0, hV, hE, EIX, mA, REP, 1, stream);
    gvp_run(nlr[l].ev1, EDGES, 0, mA, REP, nullptr, nullptr, nullptr,
            mB, REP, 1, stream);
    gvp_run(nlr[l].ev2, EDGES, 0, mB, REP, nullptr, nullptr, nullptr,
            mA, REP, 0, stream);
    ln_kernel<<<NODES, 128, 0, stream>>>(hV, mA, MSK, EIX,
                                         nlr[l].n0g, nlr[l].n0b, 1, 0);
    gvp_run(nlr[l].dh0, NODES, 0, hV, REP, nullptr, nullptr, nullptr,
            dhT, 640, 1, stream);
    gvp_run(nlr[l].dh1, NODES, 0, dhT, 640, nullptr, nullptr, nullptr,
            dhO, REP, 0, stream);
    ln_kernel<<<NODES, 128, 0, stream>>>(hV, dhO, MSK, EIX,
                                         nlr[l].n1g, nlr[l].n1b, 0, 1);
    // ---- edge layer ----
    gvp_run(el[l].ev0, EDGES, 2, nullptr, 0, hV, hE, EIX, mA, REP, 1, stream);
    gvp_run(el[l].ev1, EDGES, 0, mA, REP, nullptr, nullptr, nullptr,
            mB, REP, 1, stream);
    gvp_run(el[l].ev2, EDGES, 0, mB, REP, nullptr, nullptr, nullptr,
            mA, REP, 0, stream);
    ln_kernel<<<EDGES, 128, 0, stream>>>(hE, mA, MSK, EIX,
                                         el[l].n0g, el[l].n0b, 2, 0);
    gvp_run(el[l].dh0, EDGES, 0, hE, REP, nullptr, nullptr, nullptr,
            dhT, 640, 1, stream);
    gvp_run(el[l].dh1, EDGES, 0, dhT, 640, nullptr, nullptr, nullptr,
            dhO, REP, 0, stream);
    ln_kernel<<<EDGES, 128, 0, stream>>>(hE, dhO, MSK, EIX,
                                         el[l].n1g, el[l].n1b, 0, 2);
  }

  float* outV = (float*)d_out;
  float* outE = outV + (size_t)NODES * REP;
  gvp_run(Wout, EDGES, 0, hE, REP, nullptr, nullptr, nullptr,
          outE, REP, 0, stream);
  hipMemcpyAsync(outV, hV, (size_t)NODES * REP * sizeof(float),
                 hipMemcpyDeviceToDevice, stream);
}
